// MenghaoPointTransformerCls_81080392614603
// MI455X (gfx1250) — compile-verified, matrix operand order checked by reading, not run on hardware
//
#include <hip/hip_runtime.h>
#include <hip/hip_bf16.h>
#include <math.h>

typedef __attribute__((ext_vector_type(16))) _Float16 v16h;
typedef __attribute__((ext_vector_type(8)))  _Float16 v8h;
typedef __attribute__((ext_vector_type(8)))  float    v8f;

__device__ __forceinline__ unsigned f2ord(float f) {
  unsigned u = __float_as_uint(f);
  return (u & 0x80000000u) ? ~u : (u | 0x80000000u);
}

// ---------------------------------------------------------------------------
// WMMA GEMM on pre-converted f16 operands:
//   Out[M,N] = epilogue( A[M,K] * B^T  or  A[M,K] * B[K,N] )
// BLAYOUT=0: Bm is [N,K] row-major (K contiguous); BLAYOUT=1: Bm is [K,N]
// OUT16=1: store output as f16 (for GEMM->GEMM chains), else f32
// epilogue: acc += bias[n]; acc = acc*gamma[n]*rsqrt(1+eps)+beta[n]; ACT; +=Res
// ACT: 0 none, 1 relu, 2 leaky(0.2)
// One wave per 16x16 tile; blockDim=(32,4) -> 64x16 strip per block.
// Fast path: 4x global_load_b128 (16B of halves) + prefetch + 1 WMMA per k-step.
// ---------------------------------------------------------------------------
template<int ACT, int BLAYOUT, int OUT16>
__global__ __launch_bounds__(128)
void gemm_wmma(const _Float16* __restrict__ A, long lda, long sA,
               const _Float16* __restrict__ Bm, long ldb, long sB,
               void* Outv, long ldo, long sO,
               const float* __restrict__ Res, long ldr, long sR,
               const float* __restrict__ bias,
               const float* __restrict__ gamma,
               const float* __restrict__ beta,
               int M, int N, int K)
{
  const int b = blockIdx.z;
  A  += (long)b * sA;
  Bm += (long)b * sB;
  float*    OutF = OUT16 ? nullptr : ((float*)Outv + (long)b * sO);
  _Float16* OutH = OUT16 ? ((_Float16*)Outv + (long)b * sO) : nullptr;
  if (Res) Res += (long)b * sR;

  const int tileM = blockIdx.x * blockDim.y + threadIdx.y;
  const int tileN = blockIdx.y;
  if (tileM * 16 >= M) return;

  const int lane = threadIdx.x;
  const int half = lane >> 4;        // 0 or 1
  const int lm   = lane & 15;
  const int row  = tileM * 16 + lm;  // A row for this lane
  const int col  = tileN * 16 + lm;  // B column (output channel) for this lane

  v8f acc = {0.f, 0.f, 0.f, 0.f, 0.f, 0.f, 0.f, 0.f};

  const bool fullMN = (tileM * 16 + 16 <= M) && (tileN * 16 + 16 <= N);

  if (fullMN) {
    const _Float16* Ap = A + (long)row * lda;
    const _Float16* Bp = (BLAYOUT == 0) ? (Bm + (long)col * ldb) : (Bm + col);
    int k0 = 0;
    for (; k0 + 32 <= K; k0 += 32) {
      const int kb = k0 + half * 8;
      v8h a0 = *(const v8h*)(Ap + kb);
      v8h a1 = *(const v8h*)(Ap + kb + 16);
      v16h af = __builtin_shufflevector(a0, a1, 0, 1, 2, 3, 4, 5, 6, 7,
                                        8, 9, 10, 11, 12, 13, 14, 15);
      v16h bf;
      if (BLAYOUT == 0) {
        v8h b0 = *(const v8h*)(Bp + kb);
        v8h b1 = *(const v8h*)(Bp + kb + 16);
        bf = __builtin_shufflevector(b0, b1, 0, 1, 2, 3, 4, 5, 6, 7,
                                     8, 9, 10, 11, 12, 13, 14, 15);
      } else {
#pragma unroll
        for (int i = 0; i < 8; ++i) {
          bf[i]     = Bp[(long)(kb + i) * ldb];
          bf[i + 8] = Bp[(long)(kb + 16 + i) * ldb];
        }
      }
      if (k0 + 64 <= K) {
        __builtin_prefetch(Ap + kb + 32, 0, 1);
        if (BLAYOUT == 0) __builtin_prefetch(Bp + kb + 32, 0, 1);
      }
      acc = __builtin_amdgcn_wmma_f32_16x16x32_f16(false, af, false, bf,
                                                   (short)0, acc, false, false);
    }
    if (k0 < K) {                    // K tail (e.g. conv1's K=3)
      const int kb = k0 + half * 8;
      v16h af, bf;
#pragma unroll
      for (int i = 0; i < 8; ++i) {
        const int k1 = kb + i;
        const int k2 = kb + 16 + i;
        const int k1c = (k1 < K) ? k1 : 0;
        const int k2c = (k2 < K) ? k2 : 0;
        _Float16 a1 = Ap[k1c];
        _Float16 a2 = Ap[k2c];
        _Float16 b1, b2;
        if (BLAYOUT == 0) { b1 = Bp[k1c];              b2 = Bp[k2c]; }
        else              { b1 = Bp[(long)k1c * ldb];  b2 = Bp[(long)k2c * ldb]; }
        af[i]     = (k1 < K) ? a1 : (_Float16)0.f;
        af[i + 8] = (k2 < K) ? a2 : (_Float16)0.f;
        bf[i]     = (k1 < K) ? b1 : (_Float16)0.f;
        bf[i + 8] = (k2 < K) ? b2 : (_Float16)0.f;
      }
      acc = __builtin_amdgcn_wmma_f32_16x16x32_f16(false, af, false, bf,
                                                   (short)0, acc, false, false);
    }
  } else {
    // edge tile: clamp row/col to valid addresses, zero via selects
    const bool rowOK = row < M;
    const bool colOK = col < N;
    const int rc = rowOK ? row : (M - 1);
    const int cc = colOK ? col : (N - 1);
    const _Float16* Ap = A + (long)rc * lda;
    const _Float16* Bp = (BLAYOUT == 0) ? (Bm + (long)cc * ldb) : (Bm + cc);
    for (int k0 = 0; k0 < K; k0 += 32) {
      const int kb = k0 + half * 8;
      v16h af, bf;
#pragma unroll
      for (int i = 0; i < 8; ++i) {
        const int k1 = kb + i;
        const int k2 = kb + 16 + i;
        const int k1c = (k1 < K) ? k1 : 0;
        const int k2c = (k2 < K) ? k2 : 0;
        _Float16 a1 = Ap[k1c];
        _Float16 a2 = Ap[k2c];
        _Float16 b1, b2;
        if (BLAYOUT == 0) { b1 = Bp[k1c];              b2 = Bp[k2c]; }
        else              { b1 = Bp[(long)k1c * ldb];  b2 = Bp[(long)k2c * ldb]; }
        af[i]     = (rowOK && k1 < K) ? a1 : (_Float16)0.f;
        af[i + 8] = (rowOK && k2 < K) ? a2 : (_Float16)0.f;
        bf[i]     = (colOK && k1 < K) ? b1 : (_Float16)0.f;
        bf[i + 8] = (colOK && k2 < K) ? b2 : (_Float16)0.f;
      }
      acc = __builtin_amdgcn_wmma_f32_16x16x32_f16(false, af, false, bf,
                                                   (short)0, acc, false, false);
    }
  }

#pragma unroll
  for (int r = 0; r < 8; ++r) {
    const int m = tileM * 16 + r + half * 8;
    const int n = tileN * 16 + lm;
    if (m < M && n < N) {
      float v = acc[r];
      if (bias)  v += bias[n];
      if (gamma) v = v * (gamma[n] * rsqrtf(1.0f + 1e-5f)) + beta[n];
      if (ACT == 1) v = v > 0.f ? v : 0.f;
      else if (ACT == 2) v = v >= 0.f ? v : 0.2f * v;
      if (Res) v += Res[(long)m * ldr + n];
      if (OUT16) OutH[(long)m * ldo + n] = (_Float16)v;
      else       OutF[(long)m * ldo + n] = v;
    }
  }
}

// flat f32 -> f16
__global__ __launch_bounds__(256)
void cvt_kernel(const float* __restrict__ in, _Float16* __restrict__ out, long total)
{
  const long t = (long)blockIdx.x * blockDim.x + threadIdx.x;
  if (t >= total) return;
  out[t] = (_Float16)in[t];
}

// strided f32 -> f16 (mirrors cat slices into cat16)
__global__ __launch_bounds__(256)
void cvt_strided_kernel(const float* __restrict__ in, long ldi, long sI,
                        _Float16* __restrict__ out, long ldo, long sO,
                        int rows, int cols, long total)
{
  const long t = (long)blockIdx.x * blockDim.x + threadIdx.x;
  if (t >= total) return;
  const int c = (int)(t % cols);
  const long r = t / cols;
  const int rr = (int)(r % rows);
  const int b = (int)(r / rows);
  out[(long)b * sO + (long)rr * ldo + c] = (_Float16)in[(long)b * sI + (long)rr * ldi + c];
}

// ---------------------------------------------------------------------------
// Farthest point sampling: one block per batch, dist[] in LDS, 64-bit keyed
// argmax (ties -> smallest index, matching jnp.argmax)
// ---------------------------------------------------------------------------
__global__ __launch_bounds__(256)
void fps_kernel(const float* __restrict__ xyz, int N, int npoint,
                int* __restrict__ outIdx)
{
  const int b = blockIdx.x;
  const float* X = xyz + (long)b * N * 3;
  int* oi = outIdx + (long)b * npoint;
  extern __shared__ float dist[];
  __shared__ unsigned long long skey[256];
  __shared__ int sFar;
  const int tid = threadIdx.x;

  for (int n = tid; n < N; n += blockDim.x) dist[n] = 1e10f;
  if (tid == 0) sFar = 0;
  __syncthreads();

  for (int it = 0; it < npoint; ++it) {
    const int far = sFar;
    if (tid == 0) oi[it] = far;
    const float cx = X[far * 3 + 0];
    const float cy = X[far * 3 + 1];
    const float cz = X[far * 3 + 2];
    unsigned long long best = 0ull;
    for (int n = tid; n < N; n += blockDim.x) {
      const float dx = X[n * 3 + 0] - cx;
      const float dy = X[n * 3 + 1] - cy;
      const float dz = X[n * 3 + 2] - cz;
      const float d = dx * dx + dy * dy + dz * dz;
      const float nd = fminf(dist[n], d);
      dist[n] = nd;
      const unsigned long long key =
          ((unsigned long long)__float_as_uint(nd) << 32) |
          (unsigned)(0xFFFFFFFFu - (unsigned)n);
      if (key > best) best = key;
    }
    skey[tid] = best;
    __syncthreads();
    for (int s = blockDim.x >> 1; s > 0; s >>= 1) {
      if (tid < s && skey[tid + s] > skey[tid]) skey[tid] = skey[tid + s];
      __syncthreads();
    }
    if (tid == 0)
      sFar = (int)(0xFFFFFFFFu - (unsigned)(skey[0] & 0xFFFFFFFFu));
    __syncthreads();
  }
}

// per-row L2 norm over channels [c0, C)
__global__ __launch_bounds__(256)
void row_norm_kernel(const float* __restrict__ in, float* __restrict__ out,
                     int C, int c0, long totalRows)
{
  const long t = (long)blockIdx.x * blockDim.x + threadIdx.x;
  if (t >= totalRows) return;
  const float* p = in + t * C;
  float s = 0.f;
  for (int c = c0; c < C; ++c) { const float v = p[c]; s += v * v; }
  out[t] = sqrtf(s);
}

// ---------------------------------------------------------------------------
// For centroid s: dist row = 0.75*cos(xyz) + 0.25*cos(feat[3:]) into LDS,
// then select the nsample smallest (stable ascending) via keyed argmin.
// ---------------------------------------------------------------------------
__global__ __launch_bounds__(256)
void knn_kernel(const float* __restrict__ xyz, const float* __restrict__ newXyz,
                const float* __restrict__ feat, const float* __restrict__ newFeat,
                const float* __restrict__ normXyz, const float* __restrict__ normFeat,
                int N, int C, int S, int nsample, int* __restrict__ outIdx)
{
  const int s = blockIdx.x;
  const int b = blockIdx.y;
  const float* X  = xyz + (long)b * N * 3;
  const float* NX = newXyz + ((long)b * S + s) * 3;
  const float* F  = feat + (long)b * N * C;
  const float* NF = newFeat + ((long)b * S + s) * C;
  const float* nx = normXyz + (long)b * N;
  const float* nf = normFeat + (long)b * N;
  int* oi = outIdx + ((long)b * S + s) * nsample;

  extern __shared__ float drow[];
  __shared__ unsigned long long skey[256];
  __shared__ unsigned long long sLast;
  __shared__ float sNa, sNf;
  const int tid = threadIdx.x;

  if (tid == 0) {
    sNa = sqrtf(NX[0] * NX[0] + NX[1] * NX[1] + NX[2] * NX[2]);
    float f2 = 0.f;
    for (int c = 3; c < C; ++c) f2 += NF[c] * NF[c];
    sNf = sqrtf(f2);
  }
  __syncthreads();
  const float na = sNa, naf = sNf;

  for (int n = tid; n < N; n += blockDim.x) {
    const float d1 = NX[0] * X[n * 3 + 0] + NX[1] * X[n * 3 + 1] + NX[2] * X[n * 3 + 2];
    float d2 = 0.f;
    for (int c = 3; c < C; ++c) d2 += NF[c] * F[(long)n * C + c];
    drow[n] = 0.75f * d1 / (na * nx[n]) + 0.25f * d2 / (naf * nf[n]);
  }
  __syncthreads();

  unsigned long long last = 0ull;
  for (int k = 0; k < nsample; ++k) {
    unsigned long long best = ~0ull;
    for (int n = tid; n < N; n += blockDim.x) {
      const unsigned long long key =
          ((unsigned long long)f2ord(drow[n]) << 32) | (unsigned)n;
      if ((k == 0 || key > last) && key < best) best = key;
    }
    skey[tid] = best;
    __syncthreads();
    for (int st = blockDim.x >> 1; st > 0; st >>= 1) {
      if (tid < st && skey[tid + st] < skey[tid]) skey[tid] = skey[tid + st];
      __syncthreads();
    }
    if (tid == 0) {
      sLast = skey[0];
      oi[k] = (int)(skey[0] & 0xFFFFFFFFu);
    }
    __syncthreads();
    last = sLast;
  }
}

// out[b,s,c] = in[b, idx[b,s], c]
__global__ __launch_bounds__(256)
void gather_rows_kernel(const float* __restrict__ in, const int* __restrict__ idx,
                        float* __restrict__ out, int N, int C, int S, long total)
{
  const long t = (long)blockIdx.x * blockDim.x + threadIdx.x;
  if (t >= total) return;
  const int c = (int)(t % C);
  const long sc = t / C;
  const int s = (int)(sc % S);
  const int b = (int)(sc / S);
  const int src = idx[(long)b * S + s];
  out[t] = in[((long)b * N + src) * C + c];
}

// feat16[b,s,k,0:C] = points[idx]-center ; feat16[b,s,k,C:2C] = center
__global__ __launch_bounds__(256)
void group_feat_kernel(const float* __restrict__ feat, const float* __restrict__ newFeat,
                       const int* __restrict__ idx, _Float16* __restrict__ out,
                       int N, int C, int S, int Kn, long total)
{
  const long t = (long)blockIdx.x * blockDim.x + threadIdx.x;
  if (t >= total) return;
  const int C2 = 2 * C;
  const int c = (int)(t % C2);
  const long r = t / C2;
  const int kk = (int)(r % Kn);
  const long sr = r / Kn;
  const int s = (int)(sr % S);
  const int b = (int)(sr / S);
  const float center = newFeat[((long)b * S + s) * C + (c < C ? c : c - C)];
  float v;
  if (c < C) {
    const int src = idx[((long)b * S + s) * Kn + kk];
    v = feat[((long)b * N + src) * C + c] - center;
  } else {
    v = center;
  }
  out[t] = (_Float16)v;
}

// out[b,s,c] = max_k in[b,s,k,c] ; out row stride = ldo
__global__ __launch_bounds__(256)
void maxpool_k_kernel(const float* __restrict__ in, float* __restrict__ out,
                      int S, int Kn, int C, int ldo, long total)
{
  const long t = (long)blockIdx.x * blockDim.x + threadIdx.x;
  if (t >= total) return;
  const int c = (int)(t % C);
  const long r = t / C;
  const int s = (int)(r % S);
  const int b = (int)(r / S);
  const float* p = in + ((long)b * S + s) * (long)Kn * C + c;
  float m = p[0];
  for (int k = 1; k < Kn; ++k) m = fmaxf(m, p[(long)k * C]);
  out[((long)b * S + s) * (long)ldo + c] = m;
}

// out16[b,c] = max_n in[b,n,c]   (only consumer is lin1 GEMM)
__global__ __launch_bounds__(256)
void maxpool_pts_kernel(const float* __restrict__ in, _Float16* __restrict__ out,
                        int Np, int C, long total)
{
  const long t = (long)blockIdx.x * blockDim.x + threadIdx.x;
  if (t >= total) return;
  const int c = (int)(t % C);
  const int b = (int)(t / C);
  const float* p = in + (long)b * Np * C + c;
  float m = p[0];
  for (int n = 1; n < Np; ++n) m = fmaxf(m, p[(long)n * C]);
  out[t] = (_Float16)m;
}

// row softmax, in place; grid = (rows, B), block = Ncol (==256)
__global__ __launch_bounds__(256)
void softmax_row_kernel(float* __restrict__ att, int Ncol)
{
  const int i = blockIdx.x;
  const int b = blockIdx.y;
  float* row = att + ((long)b * gridDim.x + i) * Ncol;
  __shared__ float red[256];
  const int tid = threadIdx.x;
  const float v = (tid < Ncol) ? row[tid] : -1e30f;
  red[tid] = v;
  __syncthreads();
  for (int s = 128; s > 0; s >>= 1) {
    if (tid < s) red[tid] = fmaxf(red[tid], red[tid + s]);
    __syncthreads();
  }
  const float mx = red[0];
  __syncthreads();
  const float e = (tid < Ncol) ? __expf(v - mx) : 0.f;
  red[tid] = e;
  __syncthreads();
  for (int s = 128; s > 0; s >>= 1) {
    if (tid < s) red[tid] += red[tid + s];
    __syncthreads();
  }
  if (tid < Ncol) row[tid] = e / red[0];
}

__global__ __launch_bounds__(256)
void colsum_kernel(const float* __restrict__ att, float* __restrict__ cs,
                   int rows, int Ncol, long total)
{
  const long t = (long)blockIdx.x * blockDim.x + threadIdx.x;
  if (t >= total) return;
  const int j = (int)(t % Ncol);
  const int b = (int)(t / Ncol);
  float s = 0.f;
  for (int i = 0; i < rows; ++i) s += att[((long)b * rows + i) * Ncol + j];
  cs[t] = s;
}

__global__ __launch_bounds__(256)
void normcols_kernel(float* __restrict__ att, const float* __restrict__ cs,
                     int rows, int Ncol, long total)
{
  const long t = (long)blockIdx.x * blockDim.x + threadIdx.x;
  if (t >= total) return;
  const int j = (int)(t % Ncol);
  const long r = t / Ncol;
  const int b = (int)(r / rows);
  att[t] /= (1e-9f + cs[(long)b * Ncol + j]);
}

// out16[b,j,i] = in[b,i,j]  (only consumer is x_r GEMM A operand)
__global__ __launch_bounds__(256)
void transpose_kernel(const float* __restrict__ in, _Float16* __restrict__ out,
                      int rows, int cols, long total)
{
  const long t = (long)blockIdx.x * blockDim.x + threadIdx.x;
  if (t >= total) return;
  const int i = (int)(t % rows);
  const long r = t / rows;
  const int j = (int)(r % cols);
  const int b = (int)(r / cols);
  out[t] = (_Float16)in[((long)b * rows + i) * cols + j];
}

// out16 = A - B  (only consumer is t GEMM A operand)
__global__ __launch_bounds__(256)
void sub_kernel(const float* __restrict__ A, long lda, long sA,
                const float* __restrict__ Bm, long ldb, long sB,
                _Float16* __restrict__ out, long ldo, long sO,
                int rows, int cols, long total)
{
  const long t = (long)blockIdx.x * blockDim.x + threadIdx.x;
  if (t >= total) return;
  const int c = (int)(t % cols);
  const long r = t / cols;
  const int rr = (int)(r % rows);
  const int b = (int)(r / rows);
  out[(long)b * sO + (long)rr * ldo + c] =
      (_Float16)(A[(long)b * sA + (long)rr * lda + c] -
                 Bm[(long)b * sB + (long)rr * ldb + c]);
}

// ---------------------------------------------------------------------------
// Host-side helpers
// ---------------------------------------------------------------------------
static inline unsigned ceilDivU(long a, long b) { return (unsigned)((a + b - 1) / b); }

static void launch_gemm(hipStream_t st, int act, int bkn, int out16,
                        const _Float16* A, long lda, long sA,
                        const _Float16* Bm, long ldb, long sB,
                        void* Out, long ldo, long sO,
                        const float* Res, long ldr, long sR,
                        const float* bias, const float* gamma, const float* beta,
                        int M, int N, int K, int batch)
{
  dim3 blk(32, 4);
  dim3 grd(ceilDivU(M, 64), ceilDivU(N, 16), (unsigned)batch);
#define GL(a, bb, oo)                                                           \
  gemm_wmma<a, bb, oo><<<grd, blk, 0, st>>>(A, lda, sA, Bm, ldb, sB, Out, ldo,  \
                                            sO, Res, ldr, sR, bias, gamma, beta, M, N, K)
  const int sel = act * 4 + bkn * 2 + out16;
  switch (sel) {
    case 0:  GL(0, 0, 0); break;
    case 1:  GL(0, 0, 1); break;
    case 2:  GL(0, 1, 0); break;
    case 3:  GL(0, 1, 1); break;
    case 4:  GL(1, 0, 0); break;
    case 5:  GL(1, 0, 1); break;
    case 6:  GL(1, 1, 0); break;
    case 7:  GL(1, 1, 1); break;
    case 8:  GL(2, 0, 0); break;
    case 9:  GL(2, 0, 1); break;
    case 10: GL(2, 1, 0); break;
    case 11: GL(2, 1, 1); break;
  }
#undef GL
}

extern "C" void kernel_launch(void* const* d_in, const int* in_sizes, int n_in,
                              void* d_out, int out_size, void* d_ws, size_t ws_size,
                              hipStream_t stream)
{
  (void)in_sizes; (void)out_size; (void)ws_size;
  if (n_in < 65) return;

  const float* P[72];
  for (int i = 0; i < n_in && i < 72; ++i) P[i] = (const float*)d_in[i];

  const int B = 32, N1 = 4096, C1 = 64, S1 = 512, KNN = 32;
  const int Cl0 = 128, S2 = 256, Cl1 = 256;

  const float* x       = P[0];
  const float *bn1_g = P[3], *bn1_b = P[4], *bn2_g = P[5], *bn2_b = P[6];
  const float *l0_g1 = P[9], *l0_b1 = P[10], *l0_g2 = P[11], *l0_b2 = P[12];
  const float *l1_g1 = P[15], *l1_b1 = P[16], *l1_g2 = P[17], *l1_b2 = P[18];
  const float *sag1 = P[21], *sab1 = P[22], *sag2 = P[23], *sab2 = P[24];
  const float *fuse_g = P[54], *fuse_b = P[55];
  const float *bn6_g = P[57], *bn6_b = P[58];
  const float *lin2_b = P[60], *bn7_g = P[61], *bn7_b = P[62];
  const float *lin3_b = P[64];

  // workspace allocator (256B-aligned slabs)
  char* wp = (char*)d_ws;
  auto allocF = [&](long n) {
    float* p = (float*)wp;
    wp += ((n * 4 + 255) / 256) * 256;
    return p;
  };
  auto allocH = [&](long n) {
    _Float16* p = (_Float16*)wp;
    wp += ((n * 2 + 255) / 256) * 256;
    return p;
  };
  auto allocI = [&](long n) {
    int* p = (int*)wp;
    wp += ((n * 4 + 255) / 256) * 256;
    return p;
  };

  // f32 buffers
  float* h2       = allocF((long)B * N1 * C1);
  float* normXyz  = allocF((long)B * N1);
  float* normFeat = allocF((long)B * N1);
  float* newXyz1  = allocF((long)B * S1 * 3);
  float* newPts1  = allocF((long)B * S1 * C1);
  float* bigA     = allocF((long)B * S1 * KNN * Cl0);   // == B*S2*KNN*Cl1
  float* feat0    = allocF((long)B * S1 * Cl0);
  float* normXyz2 = allocF((long)B * S1);
  float* normFeat2= allocF((long)B * S1);
  float* newXyz2  = allocF((long)B * S2 * 3);
  float* newPts2  = allocF((long)B * S2 * Cl0);
  float* cat      = allocF((long)B * S2 * 1280);
  float* h_sa2    = allocF((long)B * S2 * Cl1);
  float* att      = allocF((long)B * S2 * S2);
  float* xrBuf    = allocF((long)B * S2 * Cl1);
  float* csum     = allocF((long)B * S2);
  float* fbuf     = allocF((long)B * S2 * 1024);
  int*   fpsIdx1  = allocI((long)B * S1);
  int*   knnIdx1  = allocI((long)B * S1 * KNN);
  int*   fpsIdx2  = allocI((long)B * S2);
  int*   knnIdx2  = allocI((long)B * S2 * KNN);

  // f16 activation buffers
  _Float16* x16    = allocH((long)B * N1 * 3);
  _Float16* h1h    = allocH((long)B * N1 * C1);
  _Float16* bigAh  = allocH((long)B * S1 * KNN * Cl0);
  _Float16* bigBh  = allocH((long)B * S1 * KNN * Cl0);
  _Float16* cat16  = allocH((long)B * S2 * 1280);
  _Float16* h_sah  = allocH((long)B * S2 * Cl1);
  _Float16* h_sa2h = allocH((long)B * S2 * Cl1);
  _Float16* qkh    = allocH((long)B * S2 * 64);
  _Float16* attTh  = allocH((long)B * S2 * S2);
  _Float16* vh     = allocH((long)B * S2 * Cl1);
  _Float16* xmrh   = allocH((long)B * S2 * Cl1);
  _Float16* gmaxh  = allocH((long)B * 1024);
  _Float16* g1h    = allocH((long)B * 512);
  _Float16* g2h    = allocH((long)B * 256);

  // f16 weight buffers
  _Float16* wConv1 = allocH(64 * 3);
  _Float16* wConv2 = allocH(64 * 64);
  _Float16* wL0c1  = allocH(128 * 128);
  _Float16* wL0c2  = allocH(128 * 128);
  _Float16* wL1c1  = allocH(256 * 256);
  _Float16* wL1c2  = allocH(256 * 256);
  _Float16* wSac1  = allocH(256 * 256);
  _Float16* wSac2  = allocH(256 * 256);
  _Float16* wQk[4], *wV[4], *wT[4];
  for (int i = 0; i < 4; ++i) {
    wQk[i] = allocH(64 * 256);
    wV[i]  = allocH(256 * 256);
    wT[i]  = allocH(256 * 256);
  }
  _Float16* wFuse = allocH(1024 * 1280);
  _Float16* wLin1 = allocH(512 * 1024);
  _Float16* wLin2 = allocH(256 * 512);
  _Float16* wLin3 = allocH(40 * 256);

  auto egrid = [](long total) { return dim3(ceilDivU(total, 256)); };
  auto cvt = [&](const float* src, _Float16* dst, long n) {
    cvt_kernel<<<egrid(n), 256, 0, stream>>>(src, dst, n);
  };

  // ---- one-time weight + input conversion to f16 ----
  cvt(P[1], wConv1, 64 * 3);
  cvt(P[2], wConv2, 64 * 64);
  cvt(P[7], wL0c1, 128 * 128);
  cvt(P[8], wL0c2, 128 * 128);
  cvt(P[13], wL1c1, 256 * 256);
  cvt(P[14], wL1c2, 256 * 256);
  cvt(P[19], wSac1, 256 * 256);
  cvt(P[20], wSac2, 256 * 256);
  for (int i = 0; i < 4; ++i) {
    cvt(P[25 + 7 * i], wQk[i], 64 * 256);
    cvt(P[26 + 7 * i], wV[i], 256 * 256);
    cvt(P[28 + 7 * i], wT[i], 256 * 256);
  }
  cvt(P[53], wFuse, 1024 * 1280);
  cvt(P[56], wLin1, 512 * 1024);
  cvt(P[59], wLin2, 256 * 512);
  cvt(P[63], wLin3, 40 * 256);
  cvt(x, x16, (long)B * N1 * 3);

  // h1h = relu(bn1(x @ conv1^T)) [f16] ; h2 = relu(bn2(h1 @ conv2^T)) [f32]
  launch_gemm(stream, 1, 0, 1, x16, 3, (long)N1 * 3, wConv1, 3, 0,
              h1h, C1, (long)N1 * C1, nullptr, 0, 0, nullptr, bn1_g, bn1_b,
              N1, C1, 3, B);
  launch_gemm(stream, 1, 0, 0, h1h, C1, (long)N1 * C1, wConv2, C1, 0,
              h2, C1, (long)N1 * C1, nullptr, 0, 0, nullptr, bn2_g, bn2_b,
              N1, C1, C1, B);

  // ---- stage 1: sample_and_group(512, 32) + local_op l0 ----
  row_norm_kernel<<<egrid((long)B * N1), 256, 0, stream>>>(x, normXyz, 3, 0, (long)B * N1);
  row_norm_kernel<<<egrid((long)B * N1), 256, 0, stream>>>(h2, normFeat, C1, 3, (long)B * N1);
  fps_kernel<<<B, 256, N1 * sizeof(float), stream>>>(x, N1, S1, fpsIdx1);
  gather_rows_kernel<<<egrid((long)B * S1 * 3), 256, 0, stream>>>(
      x, fpsIdx1, newXyz1, N1, 3, S1, (long)B * S1 * 3);
  gather_rows_kernel<<<egrid((long)B * S1 * C1), 256, 0, stream>>>(
      h2, fpsIdx1, newPts1, N1, C1, S1, (long)B * S1 * C1);
  knn_kernel<<<dim3(S1, B), 256, N1 * sizeof(float), stream>>>(
      x, newXyz1, h2, newPts1, normXyz, normFeat, N1, C1, S1, KNN, knnIdx1);
  group_feat_kernel<<<egrid((long)B * S1 * KNN * Cl0), 256, 0, stream>>>(
      h2, newPts1, knnIdx1, bigAh, N1, C1, S1, KNN, (long)B * S1 * KNN * Cl0);

  {
    const int M = S1 * KNN;
    launch_gemm(stream, 1, 0, 1, bigAh, Cl0, (long)M * Cl0, wL0c1, Cl0, 0,
                bigBh, Cl0, (long)M * Cl0, nullptr, 0, 0, nullptr, l0_g1, l0_b1,
                M, Cl0, Cl0, B);
    launch_gemm(stream, 1, 0, 0, bigBh, Cl0, (long)M * Cl0, wL0c2, Cl0, 0,
                bigA, Cl0, (long)M * Cl0, nullptr, 0, 0, nullptr, l0_g2, l0_b2,
                M, Cl0, Cl0, B);
  }
  maxpool_k_kernel<<<egrid((long)B * S1 * Cl0), 256, 0, stream>>>(
      bigA, feat0, S1, KNN, Cl0, Cl0, (long)B * S1 * Cl0);

  // ---- stage 2: sample_and_group(256, 32) + local_op l1 ----
  row_norm_kernel<<<egrid((long)B * S1), 256, 0, stream>>>(newXyz1, normXyz2, 3, 0, (long)B * S1);
  row_norm_kernel<<<egrid((long)B * S1), 256, 0, stream>>>(feat0, normFeat2, Cl0, 3, (long)B * S1);
  fps_kernel<<<B, 256, S1 * sizeof(float), stream>>>(newXyz1, S1, S2, fpsIdx2);
  gather_rows_kernel<<<egrid((long)B * S2 * 3), 256, 0, stream>>>(
      newXyz1, fpsIdx2, newXyz2, S1, 3, S2, (long)B * S2 * 3);
  gather_rows_kernel<<<egrid((long)B * S2 * Cl0), 256, 0, stream>>>(
      feat0, fpsIdx2, newPts2, S1, Cl0, S2, (long)B * S2 * Cl0);
  knn_kernel<<<dim3(S2, B), 256, S1 * sizeof(float), stream>>>(
      newXyz1, newXyz2, feat0, newPts2, normXyz2, normFeat2, S1, Cl0, S2, KNN, knnIdx2);
  group_feat_kernel<<<egrid((long)B * S2 * KNN * Cl1), 256, 0, stream>>>(
      feat0, newPts2, knnIdx2, bigAh, S1, Cl0, S2, KNN, (long)B * S2 * KNN * Cl1);

  {
    const int M = S2 * KNN;
    launch_gemm(stream, 1, 0, 1, bigAh, Cl1, (long)M * Cl1, wL1c1, Cl1, 0,
                bigBh, Cl1, (long)M * Cl1, nullptr, 0, 0, nullptr, l1_g1, l1_b1,
                M, Cl1, Cl1, B);
    launch_gemm(stream, 1, 0, 0, bigBh, Cl1, (long)M * Cl1, wL1c2, Cl1, 0,
                bigA, Cl1, (long)M * Cl1, nullptr, 0, 0, nullptr, l1_g2, l1_b2,
                M, Cl1, Cl1, B);
  }
  // feature_1 -> cat columns [1024:1280), plus f16 mirror
  maxpool_k_kernel<<<egrid((long)B * S2 * Cl1), 256, 0, stream>>>(
      bigA, cat + 1024, S2, KNN, Cl1, 1280, (long)B * S2 * Cl1);
  cvt_strided_kernel<<<egrid((long)B * S2 * Cl1), 256, 0, stream>>>(
      cat + 1024, 1280, (long)S2 * 1280, cat16 + 1024, 1280, (long)S2 * 1280,
      S2, Cl1, (long)B * S2 * Cl1);

  // ---- sa_c1 / sa_c2 ----
  launch_gemm(stream, 1, 0, 1, cat16 + 1024, 1280, (long)S2 * 1280, wSac1, Cl1, 0,
              h_sah, Cl1, (long)S2 * Cl1, nullptr, 0, 0, nullptr, sag1, sab1,
              S2, Cl1, Cl1, B);
  launch_gemm(stream, 1, 0, 0, h_sah, Cl1, (long)S2 * Cl1, wSac2, Cl1, 0,
              h_sa2, Cl1, (long)S2 * Cl1, nullptr, 0, 0, nullptr, sag2, sab2,
              S2, Cl1, Cl1, B);
  cvt(h_sa2, h_sa2h, (long)B * S2 * Cl1);

  // ---- SA layers ----
  auto run_sa = [&](const float* xinF, const _Float16* xinH, long ldx,
                    float* outF, _Float16* outH, int i) {
    const float* v_b  = P[27 + 7 * i];
    const float* t_b  = P[29 + 7 * i];
    const float* bn_g = P[30 + 7 * i];
    const float* bn_b = P[31 + 7 * i];
    const long sx = (long)S2 * ldx;
    const long sq = (long)S2 * 64;
    const long sm = (long)S2 * S2;   // 65536
    // qk = x @ qk_w^T  [f16]
    launch_gemm(stream, 0, 0, 1, xinH, ldx, sx, wQk[i], Cl1, 0,
                qkh, 64, sq, nullptr, 0, 0, nullptr, nullptr, nullptr,
                S2, 64, Cl1, B);
    // energy = qk @ qk^T  [f32]
    launch_gemm(stream, 0, 0, 0, qkh, 64, sq, qkh, 64, sq,
                att, S2, sm, nullptr, 0, 0, nullptr, nullptr, nullptr,
                S2, S2, 64, B);
    softmax_row_kernel<<<dim3(S2, B), 256, 0, stream>>>(att, S2);
    colsum_kernel<<<egrid((long)B * S2), 256, 0, stream>>>(att, csum, S2, S2, (long)B * S2);
    normcols_kernel<<<egrid((long)B * sm), 256, 0, stream>>>(att, csum, S2, S2, (long)B * sm);
    transpose_kernel<<<egrid((long)B * sm), 256, 0, stream>>>(att, attTh, S2, S2, (long)B * sm);
    // v = x @ v_w^T + v_b  [f16]
    launch_gemm(stream, 0, 0, 1, xinH, ldx, sx, wV[i], Cl1, 0,
                vh, Cl1, (long)S2 * Cl1, nullptr, 0, 0, v_b, nullptr, nullptr,
                S2, Cl1, Cl1, B);
    // x_r = att^T @ v  [f32]  (B is [K,N])
    launch_gemm(stream, 0, 1, 0, attTh, S2, sm, vh, Cl1, (long)S2 * Cl1,
                xrBuf, Cl1, (long)S2 * Cl1, nullptr, 0, 0, nullptr, nullptr, nullptr,
                S2, Cl1, S2, B);
    // xmr = x - x_r  [f16]
    sub_kernel<<<egrid((long)B * S2 * Cl1), 256, 0, stream>>>(
        xinF, ldx, sx, xrBuf, Cl1, (long)S2 * Cl1,
        xmrh, Cl1, (long)S2 * Cl1, S2, Cl1, (long)B * S2 * Cl1);
    // out = x + relu(bn((xmr @ t_w^T) + t_b))  [f32 into cat slice]
    launch_gemm(stream, 1, 0, 0, xmrh, Cl1, (long)S2 * Cl1, wT[i], Cl1, 0,
                outF, 1280, (long)S2 * 1280, xinF, ldx, sx, t_b, bn_g, bn_b,
                S2, Cl1, Cl1, B);
    // mirror the new slice into cat16
    cvt_strided_kernel<<<egrid((long)B * S2 * Cl1), 256, 0, stream>>>(
        outF, 1280, (long)S2 * 1280, outH, 1280, (long)S2 * 1280,
        S2, Cl1, (long)B * S2 * Cl1);
  };

  run_sa(h_sa2, h_sa2h, Cl1, cat + 0, cat16 + 0, 0);         // x1
  run_sa(cat + 0, cat16 + 0, 1280, cat + 256, cat16 + 256, 1);   // x2
  run_sa(cat + 256, cat16 + 256, 1280, cat + 512, cat16 + 512, 2); // x3
  run_sa(cat + 512, cat16 + 512, 1280, cat + 768, cat16 + 768, 3); // x4

  // ---- fuse (leaky 0.2 after bn), global max-pool, MLP head ----
  launch_gemm(stream, 2, 0, 0, cat16, 1280, (long)S2 * 1280, wFuse, 1280, 0,
              fbuf, 1024, (long)S2 * 1024, nullptr, 0, 0, nullptr, fuse_g, fuse_b,
              S2, 1024, 1280, B);
  maxpool_pts_kernel<<<egrid((long)B * 1024), 256, 0, stream>>>(
      fbuf, gmaxh, S2, 1024, (long)B * 1024);

  launch_gemm(stream, 1, 0, 1, gmaxh, 1024, 0, wLin1, 1024, 0,
              g1h, 512, 0, nullptr, 0, 0, nullptr, bn6_g, bn6_b,
              B, 512, 1024, 1);
  launch_gemm(stream, 1, 0, 1, g1h, 512, 0, wLin2, 512, 0,
              g2h, 256, 0, nullptr, 0, 0, lin2_b, bn7_g, bn7_b,
              B, 256, 512, 1);
  launch_gemm(stream, 0, 0, 0, g2h, 256, 0, wLin3, 256, 0,
              (float*)d_out, 40, 0, nullptr, 0, 0, lin3_b, nullptr, nullptr,
              B, 40, 256, 1);
}
